// SeqAttnContext_20023137534646
// MI455X (gfx1250) — matrix operand order repaired and checked by reading, not compiled, not run on hardware
//
#include <hip/hip_runtime.h>
#include <hip/hip_bf16.h>

// ---------------------------------------------------------------------------
// SeqAttnContext for MI455X (gfx1250, wave32, WMMA bf16, async-to-LDS)
//   contexts = softmax(mask(relu(xW^T+b) @ relu(yW^T+b)^T)) @ y
// ---------------------------------------------------------------------------

typedef __attribute__((ext_vector_type(16))) __bf16 bf16x16;
typedef __attribute__((ext_vector_type(8)))  __bf16 bf16x8;
typedef __attribute__((ext_vector_type(8)))  float  f32x8;

constexpr int Bn  = 16;
constexpr int XLn = 2048;
constexpr int YLn = 2048;
constexpr int Dn  = 1024;
constexpr int DP  = Dn + 8;     // padded LDS row (bf16 elems): +16B kills bank conflicts

union FragB16 { bf16x16 v; bf16x8 h[2]; __bf16 e[16]; };

__device__ inline f32x8 wmma_bf16(const FragB16& a, const FragB16& b, f32x8 c) {
  // 8 args: (neg_a, A, neg_b, B, c_mod, C, reuse_a, reuse_b)
  return __builtin_amdgcn_wmma_f32_16x16x32_bf16(false, a.v, false, b.v,
                                                 (short)0, c, false, false);
}

// ---------------------------------------------------------------------------
// Kernel 1: P[m,n] = relu( sum_k X[m,k]*W[n,k] + bias[n] )  -> bf16
// X tile (16x1024) converted to bf16 once per block into LDS (shared A-frags).
// B-frag: lane(0..15)=col N (= W row), 16 contiguous K per half-wave, cvt in reg.
// ---------------------------------------------------------------------------
__global__ __launch_bounds__(256)
void proj_kernel(const float* __restrict__ X, const float* __restrict__ W,
                 const float* __restrict__ bias, __bf16* __restrict__ P) {
  __shared__ __bf16 xTile[16][DP];               // ~32KB
  const int t    = threadIdx.x;
  const int lane = t & 31;
  const int wave = t >> 5;
  const int lo = lane & 15;
  const int hi = lane >> 4;
  const int m0 = blockIdx.x * 16;
  const int n0 = blockIdx.y * 128 + wave * 16;

  // Stage + convert X tile: 16384 floats as 4096 float4s, 16 per thread.
  const float4* xb4 = (const float4*)(X + (size_t)m0 * Dn);
#pragma unroll
  for (int i = 0; i < 16; ++i) {
    const int c   = i * 256 + t;                 // float4 index
    const int row = c >> 8;                      // 256 float4s per row
    const int col = (c & 255) * 4;
    const float4 v = xb4[c];
    __bf16* d = &xTile[row][col];
    d[0] = (__bf16)v.x; d[1] = (__bf16)v.y; d[2] = (__bf16)v.z; d[3] = (__bf16)v.w;
  }
  __syncthreads();

  const float* pb = W + (size_t)(n0 + lo) * Dn + hi * 16;  // B: 16 contiguous K

  f32x8 c = {};
  for (int k0 = 0; k0 < Dn; k0 += 32) {
    FragB16 a, b;
    a.h[0] = *(const bf16x8*)(&xTile[lo][k0 + hi * 16]);
    a.h[1] = *(const bf16x8*)(&xTile[lo][k0 + hi * 16 + 8]);
#pragma unroll
    for (int i = 0; i < 16; ++i) b.e[i] = (__bf16)pb[k0 + i];
    c = wmma_bf16(a, b, c);
  }

  const float bv = bias[n0 + lo];
#pragma unroll
  for (int g = 0; g < 8; ++g) {                  // C layout: row = g + 8*hi, col = lo
    float v = c[g] + bv;
    v = v > 0.0f ? v : 0.0f;
    P[(size_t)(m0 + g + 8 * hi) * Dn + n0 + lo] = (__bf16)v;
  }
}

// ---------------------------------------------------------------------------
// Kernel 2: yT[b][d][l] = (bf16) y[b][l][d]   (coalesced both sides via LDS)
// ---------------------------------------------------------------------------
__global__ __launch_bounds__(256)
void transpose_cvt(const float* __restrict__ y, __bf16* __restrict__ yT) {
  const int b  = blockIdx.z;
  const int l0 = blockIdx.x * 64;
  const int d0 = blockIdx.y * 64;
  __shared__ __bf16 tile[64][72];                // [d_local][l_local], padded
  const int t  = threadIdx.x;
  const int tx = t & 63;
  const int ty = t >> 6;                         // 0..3
#pragma unroll
  for (int i = 0; i < 16; ++i) {
    const int l = ty + 4 * i;
    tile[tx][l] = (__bf16)y[((size_t)b * YLn + l0 + l) * Dn + d0 + tx];
  }
  __syncthreads();
#pragma unroll
  for (int i = 0; i < 16; ++i) {
    const int d = ty + 4 * i;
    yT[((size_t)b * Dn + d0 + d) * YLn + l0 + tx] = tile[d][tx];
  }
}

// ---------------------------------------------------------------------------
// Kernel 3: fused masked attention. One workgroup = (batch, 16 x-rows).
// xp tile staged into LDS via GLOBAL_LOAD_ASYNC_TO_LDS_B128 (ASYNCcnt).
// Pass 1: masked row max over YL (S tiles via WMMA, shuffle + LDS reduce).
// Pass 2: recompute S, p = exp(s-m) -> LDS P-tile; each wave accumulates its
//         128-wide D chunk of the context via WMMA from pre-transposed yT.
// ---------------------------------------------------------------------------
__global__ __launch_bounds__(256)
void attn_kernel(const __bf16* __restrict__ xp, const __bf16* __restrict__ yp,
                 const __bf16* __restrict__ yT, const unsigned char* __restrict__ ymask,
                 float* __restrict__ out) {
  const int t    = threadIdx.x;
  const int lane = t & 31;
  const int wave = t >> 5;
  const int lo = lane & 15;
  const int hi = lane >> 4;
  const int b  = blockIdx.x / (XLn / 16);
  const int m0 = (blockIdx.x % (XLn / 16)) * 16;

  __shared__ __bf16 xpTile[16][DP];              // ~32KB, A-frags for S
  __shared__ __bf16 pTile[16][136];              // padded 16x128 P tile
  __shared__ float  red[8][16];
  __shared__ float  mrow[16];
  __shared__ float  lrow[16];

  const __bf16* xpb = xp + ((size_t)b * XLn + m0) * Dn;   // 32KB contiguous tile
  const __bf16* ypb = yp + (size_t)b * YLn * Dn;
  const __bf16* yTb = yT + (size_t)b * Dn * YLn;
  const unsigned char* mk = ymask + (size_t)b * YLn;

  // ---- async-copy xp tile (2048 x 16B chunks) into padded LDS rows ----
  {
    const unsigned ldsBase = (unsigned)(uintptr_t)(&xpTile[0][0]);
#pragma unroll
    for (int i = 0; i < 8; ++i) {
      const int c   = t + i * 256;               // chunk index 0..2047
      const int row = c >> 7;                    // 128 chunks per row
      const int col = c & 127;
      const unsigned ldsoff = ldsBase + (unsigned)(row * (DP * 2) + col * 16);
      const unsigned long long g =
          (unsigned long long)(uintptr_t)((const char*)xpb + (size_t)c * 16);
      asm volatile("global_load_async_to_lds_b128 %0, %1, off"
                   : : "v"(ldsoff), "v"(g) : "memory");
    }
    asm volatile("s_wait_asynccnt 0x0" ::: "memory");
  }
  __syncthreads();

  // ---------------- Pass 1: row max --------------------------------------
  float vmax[8];
#pragma unroll
  for (int g = 0; g < 8; ++g) vmax[g] = -3.0e38f;

  for (int tt = 0; tt < 16; ++tt) {
    const int yc0 = (wave + 8 * tt) * 16;
    const __bf16* pb = ypb + (size_t)(yc0 + lo) * Dn + hi * 16;
    if (tt < 15)                                 // prefetch next y_proj block
      __builtin_prefetch(pb + (size_t)128 * Dn, 0, 1);
    f32x8 s = {};
    for (int k0 = 0; k0 < Dn; k0 += 32) {
      FragB16 a, bb;
      a.h[0]  = *(const bf16x8*)(&xpTile[lo][k0 + hi * 16]);
      a.h[1]  = *(const bf16x8*)(&xpTile[lo][k0 + hi * 16 + 8]);
      bb.h[0] = *(const bf16x8*)(pb + k0);
      bb.h[1] = *(const bf16x8*)(pb + k0 + 8);
      s = wmma_bf16(a, bb, s);
    }
    const bool masked = mk[yc0 + lo] != 0;
#pragma unroll
    for (int g = 0; g < 8; ++g) {
      const float v = masked ? -__builtin_inff() : s[g];
      vmax[g] = fmaxf(vmax[g], v);
    }
  }
#pragma unroll
  for (int g = 0; g < 8; ++g)
    for (int off = 1; off < 16; off <<= 1)
      vmax[g] = fmaxf(vmax[g], __shfl_xor(vmax[g], off, 32));
  if (lo == 0) {
#pragma unroll
    for (int g = 0; g < 8; ++g) red[wave][g + 8 * hi] = vmax[g];
  }
  __syncthreads();
  if (t < 16) {
    float m = red[0][t];
    for (int w = 1; w < 8; ++w) m = fmaxf(m, red[w][t]);
    mrow[t] = m;
  }
  __syncthreads();
  float rm[8];
#pragma unroll
  for (int g = 0; g < 8; ++g) rm[g] = mrow[g + 8 * hi];

  // ---------------- Pass 2: exp + context --------------------------------
  float ls[8];
#pragma unroll
  for (int g = 0; g < 8; ++g) ls[g] = 0.0f;
  f32x8 ctx[8];
#pragma unroll
  for (int ns = 0; ns < 8; ++ns) ctx[ns] = (f32x8){};
  const int nc0 = wave * 128;                    // this wave's D chunk

  for (int j = 0; j < YLn / 128; ++j) {
    const int yb0 = j * 128;
    const int yc0 = yb0 + wave * 16;
    const __bf16* pb = ypb + (size_t)(yc0 + lo) * Dn + hi * 16;
    if (j < 15)
      __builtin_prefetch(pb + (size_t)128 * Dn, 0, 1);
    f32x8 s = {};
    for (int k0 = 0; k0 < Dn; k0 += 32) {
      FragB16 a, bb;
      a.h[0]  = *(const bf16x8*)(&xpTile[lo][k0 + hi * 16]);
      a.h[1]  = *(const bf16x8*)(&xpTile[lo][k0 + hi * 16 + 8]);
      bb.h[0] = *(const bf16x8*)(pb + k0);
      bb.h[1] = *(const bf16x8*)(pb + k0 + 8);
      s = wmma_bf16(a, bb, s);
    }
    const bool masked = mk[yc0 + lo] != 0;
#pragma unroll
    for (int g = 0; g < 8; ++g) {
      const float v = masked ? 0.0f : __expf(s[g] - rm[g]);
      ls[g] += v;
      pTile[g + 8 * hi][wave * 16 + lo] = (__bf16)v;   // A-major: [row][k]
    }
    __syncthreads();

    // ctx[16 x 128chunk] += P(16x128) @ Y(128 x chunk) ; K = local y rows
#pragma unroll
    for (int k0 = 0; k0 < 128; k0 += 32) {
      FragB16 a;
      a.h[0] = *(const bf16x8*)(&pTile[lo][k0 + hi * 16]);
      a.h[1] = *(const bf16x8*)(&pTile[lo][k0 + hi * 16 + 8]);
#pragma unroll
      for (int ns = 0; ns < 8; ++ns) {
        FragB16 bb;
        const __bf16* pbb =
            yTb + (size_t)(nc0 + ns * 16 + lo) * YLn + yb0 + k0 + hi * 16;
        bb.h[0] = *(const bf16x8*)(pbb);
        bb.h[1] = *(const bf16x8*)(pbb + 8);
        ctx[ns] = wmma_bf16(a, bb, ctx[ns]);
      }
    }
    __syncthreads();
  }

  // row sums across lanes, then across waves
#pragma unroll
  for (int g = 0; g < 8; ++g)
    for (int off = 1; off < 16; off <<= 1)
      ls[g] += __shfl_xor(ls[g], off, 32);
  if (lo == 0) {
#pragma unroll
    for (int g = 0; g < 8; ++g) red[wave][g + 8 * hi] = ls[g];
  }
  __syncthreads();
  if (t < 16) {
    float l = 0.0f;
    for (int w = 0; w < 8; ++w) l += red[w][t];
    lrow[t] = l;
  }
  __syncthreads();
  float inv[8];
#pragma unroll
  for (int g = 0; g < 8; ++g) inv[g] = 1.0f / lrow[g + 8 * hi];

  float* ob = out + ((size_t)b * XLn + m0) * Dn;
#pragma unroll
  for (int ns = 0; ns < 8; ++ns)
#pragma unroll
    for (int g = 0; g < 8; ++g)
      ob[(size_t)(g + 8 * hi) * Dn + nc0 + ns * 16 + lo] = ctx[ns][g] * inv[g];
}

// ---------------------------------------------------------------------------
// Launch: d_in = {x, y, y_mask, W, b}; d_out = contexts (B,XL,D) f32.
// ws: xp bf16 (64MB) | yp bf16 (64MB) | yT bf16 (64MB)
// ---------------------------------------------------------------------------
extern "C" void kernel_launch(void* const* d_in, const int* in_sizes, int n_in,
                              void* d_out, int out_size, void* d_ws, size_t ws_size,
                              hipStream_t stream) {
  (void)in_sizes; (void)n_in; (void)out_size; (void)ws_size;
  const float* x  = (const float*)d_in[0];
  const float* y  = (const float*)d_in[1];
  const unsigned char* ym = (const unsigned char*)d_in[2];
  const float* W  = (const float*)d_in[3];
  const float* bb = (const float*)d_in[4];
  float* out = (float*)d_out;

  char* ws = (char*)d_ws;
  __bf16* xp = (__bf16*)(ws);
  __bf16* yp = (__bf16*)(ws + ((size_t)64 << 20));
  __bf16* yT = (__bf16*)(ws + ((size_t)128 << 20));

  dim3 gp(Bn * XLn / 16, Dn / 128);                    // (2048, 8)
  proj_kernel<<<gp, 256, 0, stream>>>(x, W, bb, xp);
  proj_kernel<<<gp, 256, 0, stream>>>(y, W, bb, yp);

  dim3 gt(YLn / 64, Dn / 64, Bn);                      // (32, 16, 16)
  transpose_cvt<<<gt, 256, 0, stream>>>(y, yT);

  attn_kernel<<<Bn * (XLn / 16), 256, 0, stream>>>(xp, yp, yT, ym, out);
}
